// GAT_70325794505413
// MI455X (gfx1250) — compile-verified, matrix operand order checked by reading, not run on hardware
//
#include <hip/hip_runtime.h>
#include <hip/hip_bf16.h>

// ---------------- problem constants (match reference) ----------------
constexpr int kN   = 50000;
constexpr int kE   = 800000;
constexpr int kB   = 256;
constexpr int kDIN = 128;
constexpr int kDH  = 64;
constexpr int kH   = 3;
constexpr int kDE  = 16;
constexpr int kDOUT= 32;

typedef __bf16 bf16_t;
typedef __attribute__((ext_vector_type(16))) __bf16 v16bf;
typedef __attribute__((ext_vector_type(8)))  __bf16 v8bf;
typedef __attribute__((ext_vector_type(8)))  float  v8f;

union V16U { v16bf v; v8bf h[2]; };

// order-preserving float <-> uint key for atomicMax-based segment max
__device__ __forceinline__ unsigned fkey(float f) {
  unsigned u = __float_as_uint(f);
  return (u & 0x80000000u) ? ~u : (u | 0x80000000u);
}
__device__ __forceinline__ float fkey_dec(unsigned u) {
  return (u & 0x80000000u) ? __uint_as_float(u & 0x7fffffffu) : __uint_as_float(~u);
}
__device__ __forceinline__ float sigmoidf_(float x) { return 1.0f / (1.0f + __expf(-x)); }

__device__ __forceinline__ v8bf cvt8(float4 f0, float4 f1) {
  v8bf r;
  r[0] = (bf16_t)f0.x; r[1] = (bf16_t)f0.y; r[2] = (bf16_t)f0.z; r[3] = (bf16_t)f0.w;
  r[4] = (bf16_t)f1.x; r[5] = (bf16_t)f1.y; r[6] = (bf16_t)f1.z; r[7] = (bf16_t)f1.w;
  return r;
}

// ---------------- WMMA GEMM: C[M,Nc] = A[M,K](f32) * B (Bt[Nc,K] bf16) ----
// one wave per 16x16 output tile; grid = (Nc/16, M/16), block = 32
// Requires K % 16 == 0 (tail group handled by a UNIFORM scalar branch, so EXEC
// stays all-1s: no divergence around v_wmma; fragment loads are b128).
__global__ void wmma_gemm(const float* __restrict__ A,
                          const bf16_t* __restrict__ Bt,
                          const float* __restrict__ bias,
                          float* __restrict__ C,
                          bf16_t* __restrict__ Cbf,
                          int M, int Nc, int K, int relu) {
  const int lane = threadIdx.x & 31;
  const int n0 = blockIdx.x * 16;
  const int m0 = blockIdx.y * 16;
  const int hl = lane >> 4;    // half of wave
  const int ix = lane & 15;    // A-row / B-col within tile
  const float*  arow = A  + (size_t)(m0 + ix) * K;
  const bf16_t* brow = Bt + (size_t)(n0 + ix) * K;
  v8f acc = {0.f, 0.f, 0.f, 0.f, 0.f, 0.f, 0.f, 0.f};
  const v8bf zero8 = {(bf16_t)0.f, (bf16_t)0.f, (bf16_t)0.f, (bf16_t)0.f,
                      (bf16_t)0.f, (bf16_t)0.f, (bf16_t)0.f, (bf16_t)0.f};
  for (int k0 = 0; k0 < K; k0 += 32) {
    // ISA 16-bit A/B layout: lanes 0-15 hold K = k0+[0..7] and k0+[16..23];
    //                        lanes 16-31 hold K = k0+[8..15] and k0+[24..31].
    const int ka = k0 + hl * 8;
    const int kb = ka + 16;
    V16U a, b;
    {
      float4 f0 = *(const float4*)(arow + ka);
      float4 f1 = *(const float4*)(arow + ka + 4);
      a.h[0] = cvt8(f0, f1);
      b.h[0] = *(const v8bf*)(brow + ka);
    }
    if (k0 + 32 <= K) {          // uniform scalar condition (K % 16 == 0)
      float4 f2 = *(const float4*)(arow + kb);
      float4 f3 = *(const float4*)(arow + kb + 4);
      a.h[1] = cvt8(f2, f3);
      b.h[1] = *(const v8bf*)(brow + kb);
    } else {
      a.h[1] = zero8;
      b.h[1] = zero8;
    }
    acc = __builtin_amdgcn_wmma_f32_16x16x32_bf16(false, a.v, false, b.v,
                                                  (short)0, acc, false, false);
  }
  const int col = n0 + ix;
  const float bv = bias ? bias[col] : 0.0f;
#pragma unroll
  for (int v = 0; v < 8; ++v) {
    const int m = m0 + v + 8 * hl;         // D layout: VGPRv -> M=v (+8 for hi half)
    float val = acc[v] + bv;
    if (relu) val = fmaxf(val, 0.0f);
    const size_t o = (size_t)m * Nc + col;
    if (C)   C[o]   = val;
    if (Cbf) Cbf[o] = (bf16_t)val;
  }
}

// ---------------- weight prep ----------------
// W[K,N] f32 -> Wt[N,K] bf16
__global__ void prep_wt(const float* __restrict__ W, bf16_t* __restrict__ Wt, int K, int N) {
  int i = blockIdx.x * blockDim.x + threadIdx.x;
  if (i < K * N) {
    int n = i / K, k = i % K;
    Wt[(size_t)n * K + k] = (bf16_t)W[(size_t)k * N + n];
  }
}

// ve[h][k] = sum_c We[k, h*C + c] * ae[h*C + c]   (We: [64, heads*C])
__global__ void prep_ve(const float* __restrict__ We, const float* __restrict__ ae,
                        float* __restrict__ ve, int heads, int C) {
  int t = blockIdx.x * blockDim.x + threadIdx.x;
  if (t < heads * 64) {
    int h = t / 64, k = t % 64;
    float s = 0.f;
    for (int c = 0; c < C; ++c) s += We[(size_t)k * heads * C + h * C + c] * ae[h * C + c];
    ve[t] = s;
  }
}

// combined LSTM weight: Wct[256][192] bf16 (cols 0..127 = Wih row, 128..191 = Whh row)
__global__ void prep_comb(const float* __restrict__ Wih, const float* __restrict__ Whh,
                          const float* __restrict__ bih, const float* __restrict__ bhh,
                          bf16_t* __restrict__ Wct, float* __restrict__ bc) {
  int i = blockIdx.x * blockDim.x + threadIdx.x;
  if (i < 256 * 192) {
    int n = i / 192, k = i % 192;
    float v = (k < 128) ? Wih[(size_t)k * 256 + n] : Whh[(size_t)(k - 128) * 256 + n];
    Wct[(size_t)n * 192 + k] = (bf16_t)v;
    if (k == 0) bc[n] = bih[n] + bhh[n];
  }
}

// ---------------- edge attention precompute: alphaE[e][j] = e_bf[e,:] . ve_j ------
__global__ void edge_alpha_pre(const bf16_t* __restrict__ ebf, const float* __restrict__ veAll,
                               float* __restrict__ alphaE, int E) {
  __shared__ float sve[7 * 64];
  for (int i = threadIdx.x; i < 7 * 64; i += blockDim.x) sve[i] = veAll[i];
  __syncthreads();
  int e = blockIdx.x * blockDim.x + threadIdx.x;
  if (e >= E) return;
  const bf16_t* row = ebf + (size_t)e * 64;
  float r[7] = {0.f, 0.f, 0.f, 0.f, 0.f, 0.f, 0.f};
  for (int k = 0; k < 64; ++k) {
    float v = (float)row[k];
#pragma unroll
    for (int j = 0; j < 7; ++j) r[j] += v * sve[j * 64 + k];
  }
#pragma unroll
  for (int j = 0; j < 7; ++j) alphaE[(size_t)e * 7 + j] = r[j];
}

// ---------------- per-node attention scores ----------------
__global__ void node_scores(const float* __restrict__ xh, const float* __restrict__ as,
                            const float* __restrict__ ad, float* __restrict__ ssrc,
                            float* __restrict__ sdst, int N, int H, int C) {
  int t = blockIdx.x * blockDim.x + threadIdx.x;
  if (t >= N * H) return;
  int n = t / H, h = t % H;
  const float* row = xh + (size_t)n * H * C + h * C;
  float s1 = 0.f, s2 = 0.f;
  for (int c = 0; c < C; ++c) { s1 += row[c] * as[h * C + c]; s2 += row[c] * ad[h * C + c]; }
  ssrc[t] = s1; sdst[t] = s2;
}

// ---------------- edge softmax (3 passes) ----------------
__global__ void edge_alpha(const int* __restrict__ src, const int* __restrict__ dst,
                           const float* __restrict__ ssrc, const float* __restrict__ sdst,
                           const float* __restrict__ alphaE, int aeoff,
                           float* __restrict__ alpha, unsigned* __restrict__ amax,
                           int E, int H) {
  int t = blockIdx.x * blockDim.x + threadIdx.x;
  if (t >= E * H) return;
  int e = t / H, h = t % H;
  float a = ssrc[src[e] * H + h] + sdst[dst[e] * H + h] + alphaE[(size_t)e * 7 + aeoff + h];
  a = (a > 0.f) ? a : 0.2f * a;           // leaky_relu(0.2)
  alpha[t] = a;
  atomicMax(&amax[dst[e] * H + h], fkey(a));
}

__global__ void edge_exp(const int* __restrict__ dst, float* __restrict__ alpha,
                         const unsigned* __restrict__ amax, float* __restrict__ denom,
                         int E, int H) {
  int t = blockIdx.x * blockDim.x + threadIdx.x;
  if (t >= E * H) return;
  int e = t / H, h = t % H;
  int d = dst[e];
  float ex = __expf(alpha[t] - fkey_dec(amax[d * H + h]));
  alpha[t] = ex;
  atomicAdd(&denom[d * H + h], ex);
}

__global__ void edge_scatter(const int* __restrict__ src, const int* __restrict__ dst,
                             const float* __restrict__ ex, const float* __restrict__ denom,
                             const float* __restrict__ xh, float* __restrict__ outc,
                             int E, int H, int C) {
  long long t = (long long)blockIdx.x * blockDim.x + threadIdx.x;
  const int F = H * C;
  if (t >= (long long)E * F) return;
  int e = (int)(t / F), c = (int)(t % F);
  int h = c / C;
  int s = src[e], d = dst[e];
  float w = ex[(size_t)e * H + h] / (denom[(size_t)d * H + h] + 1e-16f);
  atomicAdd(&outc[(size_t)d * F + c], w * xh[(size_t)s * F + c]);
}

// ---------------- batchnorm (training stats) + ELU ----------------
__global__ void bn_stats(const float* __restrict__ x, float* __restrict__ sums, int N, int F) {
  int c = threadIdx.x;                       // blockDim.x == F
  int r0 = blockIdx.x * 256;
  int r1 = r0 + 256; if (r1 > N) r1 = N;
  float s = 0.f, q = 0.f;
  for (int r = r0; r < r1; ++r) { float v = x[(size_t)r * F + c]; s += v; q += v * v; }
  atomicAdd(&sums[c], s);
  atomicAdd(&sums[F + c], q);
}

__global__ void bn_apply(const float* __restrict__ x, const float* __restrict__ sums,
                         const float* __restrict__ g, const float* __restrict__ b,
                         float* __restrict__ y, int N, int F) {
  long long t = (long long)blockIdx.x * blockDim.x + threadIdx.x;
  if (t >= (long long)N * F) return;
  int c = (int)(t % F);
  float mu  = sums[c] / (float)N;
  float var = sums[F + c] / (float)N - mu * mu;
  float v = (x[t] - mu) * rsqrtf(var + 1e-5f) * g[c] + b[c];
  y[t] = (v > 0.f) ? v : (__expf(v) - 1.f);  // ELU
}

// ---------------- Set2Set ----------------
__global__ void s2s_build_qh(const float* __restrict__ qstar, const float* __restrict__ hB,
                             float* __restrict__ qh) {
  int t = blockIdx.x * blockDim.x + threadIdx.x;
  if (t >= 256 * 192) return;
  int b = t / 192, k = t % 192;
  qh[t] = (k < 128) ? qstar[b * 128 + k] : hB[b * 64 + (k - 128)];
}

__global__ void s2s_lstm(const float* __restrict__ gates, float* __restrict__ hB,
                         float* __restrict__ cB) {
  int t = blockIdx.x * blockDim.x + threadIdx.x;
  if (t >= 256 * 64) return;
  int b = t / 64, j = t % 64;
  const float* g4 = gates + (size_t)b * 256;
  float ig = sigmoidf_(g4[j]);
  float fg = sigmoidf_(g4[64 + j]);
  float gg = tanhf(g4[128 + j]);
  float og = sigmoidf_(g4[192 + j]);
  float c = fg * cB[t] + ig * gg;
  cB[t] = c;
  hB[t] = og * tanhf(c);
}

__global__ void s2s_att1(const float* __restrict__ h3, const int* __restrict__ batch,
                         const float* __restrict__ hB, float* __restrict__ eN,
                         unsigned* __restrict__ amaxB, int N) {
  int n = blockIdx.x * blockDim.x + threadIdx.x;
  if (n >= N) return;
  int b = batch[n];
  const float* xr = h3 + (size_t)n * 64;
  const float* q  = hB + (size_t)b * 64;
  float s = 0.f;
  for (int k = 0; k < 64; ++k) s += xr[k] * q[k];
  eN[n] = s;
  atomicMax(&amaxB[b], fkey(s));
}

__global__ void s2s_att2(const int* __restrict__ batch, float* __restrict__ eN,
                         const unsigned* __restrict__ amaxB, float* __restrict__ denB, int N) {
  int n = blockIdx.x * blockDim.x + threadIdx.x;
  if (n >= N) return;
  int b = batch[n];
  float ex = __expf(eN[n] - fkey_dec(amaxB[b]));
  eN[n] = ex;
  atomicAdd(&denB[b], ex);
}

__global__ void s2s_att3(const float* __restrict__ h3, const int* __restrict__ batch,
                         const float* __restrict__ eN, const float* __restrict__ denB,
                         float* __restrict__ r, int N) {
  long long t = (long long)blockIdx.x * blockDim.x + threadIdx.x;
  if (t >= (long long)N * 64) return;
  int n = (int)(t / 64), c = (int)(t % 64);
  int b = batch[n];
  float a = eN[n] / (denB[b] + 1e-16f);
  atomicAdd(&r[(size_t)b * 64 + c], a * h3[t]);
}

__global__ void s2s_qstar(const float* __restrict__ hB, const float* __restrict__ r,
                          float* __restrict__ qstar) {
  int t = blockIdx.x * blockDim.x + threadIdx.x;
  if (t >= 256 * 128) return;
  int b = t / 128, k = t % 128;
  qstar[t] = (k < 64) ? hB[b * 64 + k] : r[b * 64 + (k - 64)];
}

// ============================ launch ============================
extern "C" void kernel_launch(void* const* d_in, const int* in_sizes, int n_in,
                              void* d_out, int out_size, void* d_ws, size_t ws_size,
                              hipStream_t stream) {
  const float* x     = (const float*)d_in[0];
  const int*   ei    = (const int*)d_in[1];
  const float* eattr = (const float*)d_in[2];
  const int*   batch = (const int*)d_in[3];
  const float* W_ef  = (const float*)d_in[4];
  const float* b_ef  = (const float*)d_in[5];
  const float* W1 = (const float*)d_in[6],  *as1 = (const float*)d_in[7],
             *ad1 = (const float*)d_in[8],  *We1 = (const float*)d_in[9],
             *ae1 = (const float*)d_in[10];
  const float* W2 = (const float*)d_in[12], *as2 = (const float*)d_in[13],
             *ad2 = (const float*)d_in[14], *We2 = (const float*)d_in[15],
             *ae2 = (const float*)d_in[16];
  const float* W3 = (const float*)d_in[18], *as3 = (const float*)d_in[19],
             *ad3 = (const float*)d_in[20], *We3 = (const float*)d_in[21],
             *ae3 = (const float*)d_in[22];
  const float* g1 = (const float*)d_in[24], *be1 = (const float*)d_in[25];
  const float* g2 = (const float*)d_in[26], *be2 = (const float*)d_in[27];
  const float* g3 = (const float*)d_in[28], *be3 = (const float*)d_in[29];
  const float* Wih = (const float*)d_in[30], *Whh = (const float*)d_in[31];
  const float* bih = (const float*)d_in[32], *bhh = (const float*)d_in[33];
  const float* Wf1 = (const float*)d_in[34], *bf1 = (const float*)d_in[35];
  const float* Wf2 = (const float*)d_in[36], *bf2 = (const float*)d_in[37];
  const int* src = ei;
  const int* dst = ei + kE;
  float* out = (float*)d_out;

  // ---- workspace bump allocator (e_bf arena aliased with conv activations) ----
  char* p = (char*)d_ws;
  auto alloc = [&](size_t n) -> char* {
    char* r = p;
    p += (n + 255) & ~(size_t)255;
    return r;
  };
  const size_t XH_B = (size_t)kN * 192 * sizeof(float);           // 38.4 MB
  const size_t EBF_B = (size_t)kE * 64 * sizeof(bf16_t);          // 102.4 MB
  char* arena = alloc(EBF_B);                                     // > XH_B*2
  bf16_t* EBF  = (bf16_t*)arena;
  float*  XH   = (float*)arena;                                   // alias (after alphaE)
  float*  OUTC = (float*)(arena + ((XH_B + 255) & ~(size_t)255)); // alias
  float* ALPHAE = (float*)alloc((size_t)kE * 7 * sizeof(float));
  float* HACT   = (float*)alloc(XH_B);
  float* H3     = (float*)alloc((size_t)kN * 64 * sizeof(float));
  float* SS     = (float*)alloc((size_t)kN * 3 * sizeof(float));
  float* SD     = (float*)alloc((size_t)kN * 3 * sizeof(float));
  float* ALPHA  = (float*)alloc((size_t)kE * 3 * sizeof(float));
  unsigned* AMAX = (unsigned*)alloc((size_t)kN * 3 * sizeof(unsigned));
  float* DEN    = (float*)alloc((size_t)kN * 3 * sizeof(float));
  float* EN     = (float*)alloc((size_t)kN * sizeof(float));
  float* QSTAR  = (float*)alloc(256 * 128 * sizeof(float));
  float* HB     = (float*)alloc(256 * 64 * sizeof(float));
  float* CBUF   = (float*)alloc(256 * 64 * sizeof(float));
  float* QH     = (float*)alloc(256 * 192 * sizeof(float));
  float* GATES  = (float*)alloc(256 * 256 * sizeof(float));
  float* RBUF   = (float*)alloc(256 * 64 * sizeof(float));
  float* T1     = (float*)alloc(256 * 64 * sizeof(float));
  unsigned* AMAXB = (unsigned*)alloc(256 * sizeof(unsigned));
  float* DENB   = (float*)alloc(256 * sizeof(float));
  float* SUMS   = (float*)alloc(2 * 192 * sizeof(float));
  bf16_t* WEFT = (bf16_t*)alloc(64 * 16 * sizeof(bf16_t));
  bf16_t* W1T  = (bf16_t*)alloc(192 * 128 * sizeof(bf16_t));
  bf16_t* W2T  = (bf16_t*)alloc(192 * 192 * sizeof(bf16_t));
  bf16_t* W3T  = (bf16_t*)alloc(64 * 192 * sizeof(bf16_t));
  bf16_t* WCT  = (bf16_t*)alloc(256 * 192 * sizeof(bf16_t));
  bf16_t* WF1T = (bf16_t*)alloc(64 * 128 * sizeof(bf16_t));
  bf16_t* WF2T = (bf16_t*)alloc(32 * 64 * sizeof(bf16_t));
  float* VEALL = (float*)alloc(7 * 64 * sizeof(float));
  float* BC    = (float*)alloc(256 * sizeof(float));

  auto blk = [](long long n, int t) { return (unsigned)((n + t - 1) / t); };

  // ---- weight prep (bf16 transposed) ----
  prep_wt<<<blk(16 * 64, 256), 256, 0, stream>>>(W_ef, WEFT, 16, 64);
  prep_wt<<<blk(128 * 192, 256), 256, 0, stream>>>(W1, W1T, 128, 192);
  prep_wt<<<blk(192 * 192, 256), 256, 0, stream>>>(W2, W2T, 192, 192);
  prep_wt<<<blk(192 * 64, 256), 256, 0, stream>>>(W3, W3T, 192, 64);
  prep_wt<<<blk(128 * 64, 256), 256, 0, stream>>>(Wf1, WF1T, 128, 64);
  prep_wt<<<blk(64 * 32, 256), 256, 0, stream>>>(Wf2, WF2T, 64, 32);
  prep_comb<<<blk(256 * 192, 256), 256, 0, stream>>>(Wih, Whh, bih, bhh, WCT, BC);
  prep_ve<<<blk(3 * 64, 192), 192, 0, stream>>>(We1, ae1, VEALL + 0, 3, kDH);
  prep_ve<<<blk(3 * 64, 192), 192, 0, stream>>>(We2, ae2, VEALL + 192, 3, kDH);
  prep_ve<<<blk(1 * 64, 64), 64, 0, stream>>>(We3, ae3, VEALL + 384, 1, kDH);

  // ---- edge MLP: e = relu(edge_attr @ W_ef + b_ef) -> bf16, then alphaE[e][7] ----
  {
    dim3 g(64 / 16, kE / 16);
    wmma_gemm<<<g, 32, 0, stream>>>(eattr, WEFT, b_ef, nullptr, EBF, kE, 64, kDE, 1);
  }
  edge_alpha_pre<<<blk(kE, 256), 256, 0, stream>>>(EBF, VEALL, ALPHAE, kE);

  // ---- one GAT conv layer ----
  auto conv = [&](const float* Ain, int K, const bf16_t* Wt, int H, int aeoff,
                  const float* as_, const float* ad_, const float* g_, const float* be_,
                  float* Yout) {
    const int F = H * kDH;
    hipMemsetAsync(AMAX, 0, (size_t)kN * H * sizeof(unsigned), stream);
    hipMemsetAsync(DEN, 0, (size_t)kN * H * sizeof(float), stream);
    hipMemsetAsync(OUTC, 0, (size_t)kN * F * sizeof(float), stream);
    hipMemsetAsync(SUMS, 0, 2 * F * sizeof(float), stream);
    dim3 g(F / 16, kN / 16);
    wmma_gemm<<<g, 32, 0, stream>>>(Ain, Wt, nullptr, XH, nullptr, kN, F, K, 0);
    node_scores<<<blk((long long)kN * H, 256), 256, 0, stream>>>(XH, as_, ad_, SS, SD, kN, H, kDH);
    edge_alpha<<<blk((long long)kE * H, 256), 256, 0, stream>>>(src, dst, SS, SD, ALPHAE, aeoff,
                                                                ALPHA, AMAX, kE, H);
    edge_exp<<<blk((long long)kE * H, 256), 256, 0, stream>>>(dst, ALPHA, AMAX, DEN, kE, H);
    edge_scatter<<<blk((long long)kE * F, 256), 256, 0, stream>>>(src, dst, ALPHA, DEN, XH, OUTC,
                                                                  kE, H, kDH);
    bn_stats<<<blk(kN, 256), F, 0, stream>>>(OUTC, SUMS, kN, F);
    bn_apply<<<blk((long long)kN * F, 256), 256, 0, stream>>>(OUTC, SUMS, g_, be_, Yout, kN, F);
  };

  conv(x,    kDIN,     W1T, 3, 0, as1, ad1, g1, be1, HACT);
  conv(HACT, kH * kDH, W2T, 3, 3, as2, ad2, g2, be2, HACT);
  conv(HACT, kH * kDH, W3T, 1, 6, as3, ad3, g3, be3, H3);

  // ---- Set2Set (3 steps) ----
  hipMemsetAsync(QSTAR, 0, 256 * 128 * sizeof(float), stream);
  hipMemsetAsync(HB, 0, 256 * 64 * sizeof(float), stream);
  hipMemsetAsync(CBUF, 0, 256 * 64 * sizeof(float), stream);
  for (int step = 0; step < 3; ++step) {
    s2s_build_qh<<<blk(256 * 192, 256), 256, 0, stream>>>(QSTAR, HB, QH);
    {
      dim3 g(256 / 16, 256 / 16);
      wmma_gemm<<<g, 32, 0, stream>>>(QH, WCT, BC, GATES, nullptr, 256, 256, 192, 0);
    }
    s2s_lstm<<<blk(256 * 64, 256), 256, 0, stream>>>(GATES, HB, CBUF);
    hipMemsetAsync(AMAXB, 0, 256 * sizeof(unsigned), stream);
    hipMemsetAsync(DENB, 0, 256 * sizeof(float), stream);
    hipMemsetAsync(RBUF, 0, 256 * 64 * sizeof(float), stream);
    s2s_att1<<<blk(kN, 256), 256, 0, stream>>>(H3, batch, HB, EN, AMAXB, kN);
    s2s_att2<<<blk(kN, 256), 256, 0, stream>>>(batch, EN, AMAXB, DENB, kN);
    s2s_att3<<<blk((long long)kN * 64, 256), 256, 0, stream>>>(H3, batch, EN, DENB, RBUF, kN);
    s2s_qstar<<<blk(256 * 128, 256), 256, 0, stream>>>(HB, RBUF, QSTAR);
  }

  // ---- MLP head: relu(q* @ Wf1 + bf1) @ Wf2 + bf2 ----
  {
    dim3 g1d(64 / 16, 256 / 16);
    wmma_gemm<<<g1d, 32, 0, stream>>>(QSTAR, WF1T, bf1, T1, nullptr, 256, 64, 128, 1);
    dim3 g2d(32 / 16, 256 / 16);
    wmma_gemm<<<g2d, 32, 0, stream>>>(T1, WF2T, bf2, out, nullptr, 256, 32, 64, 0);
  }
}
